// CompositePulseTransformerDecoder_30442728194298
// MI455X (gfx1250) — compile-verified, matrix operand order checked by reading, not run on hardware
//
#include <hip/hip_runtime.h>
#include <hip/hip_bf16.h>
#include <math.h>

#define DD     256
#define HH     4
#define HDIM   64
#define NLAYER 12
#define MAXPOS 16
#define BB     128

typedef __attribute__((ext_vector_type(16))) _Float16 v16h;
typedef __attribute__((ext_vector_type(8)))  _Float16 v8h;
typedef __attribute__((ext_vector_type(8)))  float    v8f;

// ---------------------------------------------------------------------------
// Elementwise f32 -> f16 (one-time weight conversion; weights then live in L2)
// ---------------------------------------------------------------------------
__global__ void cptd_f32_to_f16(const float* __restrict__ src,
                                _Float16* __restrict__ dst, int n)
{
    int t = blockIdx.x * blockDim.x + threadIdx.x;
    if (t < n) dst[t] = (_Float16)src[t];
}

// vec = concat(U_real.reshape(B,64), U_imag.reshape(B,64))  -> B x 128
__global__ void cptd_build_vec(const float* __restrict__ ur,
                               const float* __restrict__ ui,
                               float* __restrict__ vec)
{
    int t = blockIdx.x * blockDim.x + threadIdx.x;
    if (t >= BB * 128) return;
    int b = t >> 7, j = t & 127;
    vec[t] = (j < 64) ? ur[b * 64 + j] : ui[b * 64 + (j - 64)];
}

// x[b,:] = tok[b,:] + pos[:]
__global__ void cptd_make_input(const float* __restrict__ tok,
                                const float* __restrict__ pos,
                                float* __restrict__ x)
{
    int t = blockIdx.x * blockDim.x + threadIdx.x;
    if (t >= BB * DD) return;
    x[t] = tok[t] + pos[t & (DD - 1)];
}

// ---------------------------------------------------------------------------
// Double-buffered chunk state: one 32-wide K chunk of A (f32 raw) and the
// four B fragments (f16 raw halves).
// ---------------------------------------------------------------------------
struct cptd_chunk {
    float4 a0, a1, a2, a3;
    v8h    blo[4], bhi[4];
};

__device__ __forceinline__ void cptd_load_chunk(const float* __restrict__ ap,
                                                const _Float16* __restrict__ wp,
                                                size_t wstride, cptd_chunk& r)
{
    r.a0 = *(const float4*)(ap);
    r.a1 = *(const float4*)(ap + 4);
    r.a2 = *(const float4*)(ap + 16);
    r.a3 = *(const float4*)(ap + 20);
#pragma unroll
    for (int nt = 0; nt < 4; ++nt) {
        const _Float16* w = wp + (size_t)nt * wstride;
        r.blo[nt] = *(const v8h*)(w);
        r.bhi[nt] = *(const v8h*)(w + 16);
    }
}

__device__ __forceinline__ void cptd_compute_chunk(const cptd_chunk& r, v8f acc[4])
{
    v16h a;
    a[0]  = (_Float16)r.a0.x; a[1]  = (_Float16)r.a0.y;
    a[2]  = (_Float16)r.a0.z; a[3]  = (_Float16)r.a0.w;
    a[4]  = (_Float16)r.a1.x; a[5]  = (_Float16)r.a1.y;
    a[6]  = (_Float16)r.a1.z; a[7]  = (_Float16)r.a1.w;
    a[8]  = (_Float16)r.a2.x; a[9]  = (_Float16)r.a2.y;
    a[10] = (_Float16)r.a2.z; a[11] = (_Float16)r.a2.w;
    a[12] = (_Float16)r.a3.x; a[13] = (_Float16)r.a3.y;
    a[14] = (_Float16)r.a3.z; a[15] = (_Float16)r.a3.w;
#pragma unroll
    for (int nt = 0; nt < 4; ++nt) {
        v16h b = __builtin_shufflevector(r.blo[nt], r.bhi[nt],
                   0,1,2,3,4,5,6,7,8,9,10,11,12,13,14,15);
        acc[nt] = __builtin_amdgcn_wmma_f32_16x16x32_f16(
            false, a, false, b, (short)0, acc[nt], false, false);
    }
}

// ---------------------------------------------------------------------------
// WMMA GEMM:  C[M,N] = A[M,K](f32) @ W[N,K](f16)^T + bias[N]   (optional ReLU)
// One wave computes a 16 x 64 strip (4 N-tiles). Software-pipelined K loop:
// chunk k+1's 12 b128 loads are issued before chunk k's 4 WMMAs, so the
// waits become "loadcnt <= in-flight" instead of 0 and L2 latency is hidden
// behind the matrix pipe. N multiple of 64 (256/768/1024), K multiple of 32.
// Fragment layout per ISA: lane&15 = row, half-wave selects K-phase:
//   frag[e] = X[r][k0+kb+e] (e<8), frag[e+8] = X[r][k0+16+kb+e], kb=8*(lane>>4)
// ---------------------------------------------------------------------------
__global__ __launch_bounds__(256)
void cptd_gemm_wmma(const float* __restrict__ A, const _Float16* __restrict__ W,
                    const float* __restrict__ bias, float* __restrict__ C,
                    int M, int N, int K, int relu)
{
    int ng    = N >> 6;                            // groups of 4 N-tiles
    int total = (M >> 4) * ng;
    int wave  = blockIdx.x * (blockDim.x >> 5) + (threadIdx.x >> 5);
    if (wave >= total) return;                     // uniform per wave

    int tm   = wave / ng;
    int tg   = wave - tm * ng;
    int lane = threadIdx.x & 31;
    int half = lane >> 4;
    int idx  = lane & 15;
    int kb   = half << 3;

    const float*    Ar = A + (size_t)(tm * 16 + idx) * K + kb;
    const _Float16* Wr = W + (size_t)(tg * 64 + idx) * K + kb;  // rows n,n+16,n+32,n+48
    const size_t    ws = (size_t)16 * K;

    v8f acc[4] = {v8f{}, v8f{}, v8f{}, v8f{}};

    cptd_chunk cur, nxt;
    cptd_load_chunk(Ar, Wr, ws, cur);              // prologue: chunk 0 in flight

    int k0 = 0;
#pragma unroll 2
    for (; k0 + 32 < K; k0 += 32) {
        cptd_load_chunk(Ar + k0 + 32, Wr + k0 + 32, ws, nxt);  // issue next chunk
        cptd_compute_chunk(cur, acc);                          // consume current
        cur = nxt;                                             // rotate (reg renames)
    }
    cptd_compute_chunk(cur, acc);                  // epilogue chunk

    // ---- epilogue: bias (+ReLU), C/D layout VGPR r -> M=r (+8 hi half), N=lane&15
#pragma unroll
    for (int nt = 0; nt < 4; ++nt) {
        int   col = tg * 64 + nt * 16 + idx;
        float bv  = bias ? bias[col] : 0.0f;
#pragma unroll
        for (int r = 0; r < 8; ++r) {
            int   row = tm * 16 + r + half * 8;
            float v   = acc[nt][r] + bv;
            if (relu) v = fmaxf(v, 0.0f);
            C[(size_t)row * N + col] = v;
        }
    }
}

// ---------------------------------------------------------------------------
// Self-attention, single new token, exact KV cache.
// One block of 64 threads per (b,h); thread d = head dim.
// ---------------------------------------------------------------------------
__global__ void cptd_attn_step(const float* __restrict__ qkv,
                               float* __restrict__ kc, float* __restrict__ vc,
                               float* __restrict__ attno, int S)
{
    int bh = blockIdx.x;
    int b  = bh / HH, h = bh - b * HH;
    int d  = threadIdx.x;                          // 0..63

    const float* qrow = qkv + (size_t)b * (3 * DD);
    float q  = qrow[h * HDIM + d];
    float kn = qrow[DD + h * HDIM + d];
    float vn = qrow[2 * DD + h * HDIM + d];

    float* kcb = kc + (size_t)bh * MAXPOS * HDIM;
    float* vcb = vc + (size_t)bh * MAXPOS * HDIM;
    int pos = S - 1;
    kcb[pos * HDIM + d] = kn;
    vcb[pos * HDIM + d] = vn;

    __shared__ float qs[HDIM];
    __shared__ float sc[MAXPOS];
    qs[d] = q;
    __syncthreads();

    if (d < S) {
        float acc = 0.0f;
        for (int j = 0; j < HDIM; ++j) acc += qs[j] * kcb[d * HDIM + j];
        sc[d] = acc * 0.125f;                      // 1/sqrt(64)
    }
    __syncthreads();

    float mx = -1e30f;
    for (int t = 0; t < S; ++t) mx = fmaxf(mx, sc[t]);
    float den = 0.0f, o = 0.0f;
    for (int t = 0; t < S; ++t) {
        float w = expf(sc[t] - mx);
        den += w;
        o   += w * vcb[t * HDIM + d];
    }
    attno[(size_t)b * DD + h * HDIM + d] = o / den;
}

// ---------------------------------------------------------------------------
// x[row,:] = LayerNorm(x[row,:] + t[row,:]) * g + b   (row length 256)
// ---------------------------------------------------------------------------
__global__ void cptd_resln(float* __restrict__ x, const float* __restrict__ t,
                           const float* __restrict__ g, const float* __restrict__ bta,
                           int M)
{
    int row = blockIdx.x * (blockDim.x >> 5) + (threadIdx.x >> 5);
    if (row >= M) return;
    int lane = threadIdx.x & 31;

    float*       xr = x + (size_t)row * DD;
    const float* tr = t + (size_t)row * DD;

    float v[8];
    float s = 0.0f;
#pragma unroll
    for (int j = 0; j < 8; ++j) {
        int c = lane + j * 32;
        v[j]  = xr[c] + tr[c];
        s    += v[j];
    }
#pragma unroll
    for (int off = 16; off > 0; off >>= 1) s += __shfl_xor(s, off, 32);
    float mean = s * (1.0f / DD);

    float var = 0.0f;
#pragma unroll
    for (int j = 0; j < 8; ++j) { float dlt = v[j] - mean; var += dlt * dlt; }
#pragma unroll
    for (int off = 16; off > 0; off >>= 1) var += __shfl_xor(var, off, 32);
    float inv = rsqrtf(var * (1.0f / DD) + 1e-5f);

#pragma unroll
    for (int j = 0; j < 8; ++j) {
        int c = lane + j * 32;
        xr[c] = (v[j] - mean) * inv * g[c] + bta[c];
    }
}

// ---------------------------------------------------------------------------
// out[b,i,p] = plow[p] + (phigh[p]-plow[p]) * sigmoid(outs[i,b,:].W[p,:] + bias[p])
// ---------------------------------------------------------------------------
__global__ void cptd_out_proj(const float* __restrict__ outs,   // 16 x B x 256
                              const float* __restrict__ W,      // 4 x 256
                              const float* __restrict__ bias,   // 4
                              const float* __restrict__ plow,
                              const float* __restrict__ phigh,
                              float* __restrict__ out)          // B x 16 x 4
{
    int t = blockIdx.x * blockDim.x + threadIdx.x;
    if (t >= BB * MAXPOS * 4) return;
    int p = t & 3;
    int i = (t >> 2) & 15;
    int b = t >> 6;
    const float* row = outs + ((size_t)i * BB + b) * DD;
    const float* wr  = W + p * DD;
    float acc = bias[p];
    for (int j = 0; j < DD; ++j) acc += row[j] * wr[j];
    float s = 1.0f / (1.0f + expf(-acc));
    out[((size_t)b * MAXPOS + i) * 4 + p] = plow[p] + (phigh[p] - plow[p]) * s;
}

// ---------------------------------------------------------------------------
static inline void launch_gemm(const float* A, const _Float16* W, const float* bias,
                               float* C, int M, int N, int K, int relu,
                               hipStream_t s)
{
    int waves  = (M / 16) * (N / 64);
    int blocks = (waves + 7) / 8;                  // 8 waves / 256-thread block
    cptd_gemm_wmma<<<blocks, 256, 0, s>>>(A, W, bias, C, M, N, K, relu);
}

static inline void launch_cvt(const float* src, _Float16* dst, size_t n, hipStream_t s)
{
    cptd_f32_to_f16<<<(int)((n + 255) / 256), 256, 0, s>>>(src, dst, (int)n);
}

extern "C" void kernel_launch(void* const* d_in, const int* in_sizes, int n_in,
                              void* d_out, int out_size, void* d_ws, size_t ws_size,
                              hipStream_t stream)
{
    const float* U_real  = (const float*)d_in[0];
    const float* U_imag  = (const float*)d_in[1];
    const float* tok_W   = (const float*)d_in[2];
    const float* tok_b   = (const float*)d_in[3];
    const float* pos_emb = (const float*)d_in[4];
    const float* sa_Wqkv = (const float*)d_in[5];
    const float* sa_bqkv = (const float*)d_in[6];
    const float* sa_Wo   = (const float*)d_in[7];
    const float* sa_bo   = (const float*)d_in[8];
    const float* ca_Wqkv = (const float*)d_in[9];
    const float* ca_bqkv = (const float*)d_in[10];
    const float* ca_Wo   = (const float*)d_in[11];
    const float* ca_bo   = (const float*)d_in[12];
    const float* lin1_W  = (const float*)d_in[13];
    const float* lin1_b  = (const float*)d_in[14];
    const float* lin2_W  = (const float*)d_in[15];
    const float* lin2_b  = (const float*)d_in[16];
    const float* ln1_g   = (const float*)d_in[17];
    const float* ln1_b   = (const float*)d_in[18];
    const float* ln2_g   = (const float*)d_in[19];
    const float* ln2_b   = (const float*)d_in[20];
    const float* ln3_g   = (const float*)d_in[21];
    const float* ln3_b   = (const float*)d_in[22];
    const float* out_W   = (const float*)d_in[23];
    const float* out_b   = (const float*)d_in[24];
    const float* plow    = (const float*)d_in[25];
    const float* phigh   = (const float*)d_in[26];
    (void)in_sizes; (void)n_in; (void)out_size; (void)ws_size;

    // ---- f16 weight region first (16B-aligned by construction)
    _Float16* wh = (_Float16*)d_ws;
    size_t ho = 0;
    _Float16* h_tokW   = wh + ho; ho += (size_t)DD * 128;
    _Float16* h_saWqkv = wh + ho; ho += (size_t)NLAYER * 3 * DD * DD;
    _Float16* h_saWo   = wh + ho; ho += (size_t)NLAYER * DD * DD;
    _Float16* h_lin1   = wh + ho; ho += (size_t)NLAYER * 4 * DD * DD;
    _Float16* h_lin2   = wh + ho; ho += (size_t)NLAYER * DD * 4 * DD;
    _Float16* h_caWqkv = wh + ho; ho += (size_t)NLAYER * 3 * DD * DD;
    _Float16* h_caWo   = wh + ho; ho += (size_t)NLAYER * DD * DD;
    ho = (ho + 7) & ~(size_t)7;

    // ---- f32 scratch region
    float* ws = (float*)(wh + ho);
    size_t o = 0;
    float* vec    = ws + o; o += (size_t)BB * 128;
    float* src    = ws + o; o += (size_t)BB * DD;
    float* x      = ws + o; o += (size_t)BB * DD;
    float* qkv    = ws + o; o += (size_t)BB * 3 * DD;
    float* attno  = ws + o; o += (size_t)BB * DD;
    float* tbuf   = ws + o; o += (size_t)BB * DD;
    float* hbuf   = ws + o; o += (size_t)BB * 4 * DD;
    float* cav    = ws + o; o += (size_t)BB * DD;
    float* ca_out = ws + o; o += (size_t)NLAYER * BB * DD;
    float* outs   = ws + o; o += (size_t)MAXPOS * BB * DD;
    float* kc     = ws + o; o += (size_t)NLAYER * BB * HH * MAXPOS * HDIM;
    float* vc     = ws + o; o += (size_t)NLAYER * BB * HH * MAXPOS * HDIM;

    // ---- one-time weight conversion to f16 (deterministic, re-done each call)
    launch_cvt(tok_W,   h_tokW,   (size_t)DD * 128,               stream);
    launch_cvt(sa_Wqkv, h_saWqkv, (size_t)NLAYER * 3 * DD * DD,   stream);
    launch_cvt(sa_Wo,   h_saWo,   (size_t)NLAYER * DD * DD,       stream);
    launch_cvt(lin1_W,  h_lin1,   (size_t)NLAYER * 4 * DD * DD,   stream);
    launch_cvt(lin2_W,  h_lin2,   (size_t)NLAYER * DD * 4 * DD,   stream);
    launch_cvt(ca_Wqkv, h_caWqkv, (size_t)NLAYER * 3 * DD * DD,   stream);
    launch_cvt(ca_Wo,   h_caWo,   (size_t)NLAYER * DD * DD,       stream);

    // ---- token projection: src = vec @ tok_W.T + tok_b  (M=128,K=128,N=256)
    cptd_build_vec<<<(BB * 128 + 255) / 256, 256, 0, stream>>>(U_real, U_imag, vec);
    launch_gemm(vec, h_tokW, tok_b, src, BB, DD, 128, 0, stream);

    // ---- cross-attention collapses (single key -> softmax == 1):
    // ca_out[l] = (src @ Wv_l.T + bv_l) @ Wo_l.T + bo_l, precomputed once.
    for (int l = 0; l < NLAYER; ++l) {
        const _Float16* Wv = h_caWqkv + ((size_t)l * 3 * DD + 2 * DD) * DD;
        const float*    bv = ca_bqkv + (size_t)l * 3 * DD + 2 * DD;
        launch_gemm(src, Wv, bv, cav, BB, DD, DD, 0, stream);
        launch_gemm(cav, h_caWo + (size_t)l * DD * DD, ca_bo + (size_t)l * DD,
                    ca_out + (size_t)l * BB * DD, BB, DD, DD, 0, stream);
    }

    // ---- autoregressive loop with exact KV caching (causal attention)
    for (int i = 0; i < MAXPOS; ++i) {
        const float* tok = (i == 0) ? src : (outs + (size_t)(i - 1) * BB * DD);
        cptd_make_input<<<(BB * DD + 255) / 256, 256, 0, stream>>>(
            tok, pos_emb + (size_t)i * DD, x);

        for (int l = 0; l < NLAYER; ++l) {
            // self-attention
            launch_gemm(x, h_saWqkv + (size_t)l * 3 * DD * DD,
                        sa_bqkv + (size_t)l * 3 * DD, qkv, BB, 3 * DD, DD, 0, stream);
            cptd_attn_step<<<BB * HH, HDIM, 0, stream>>>(
                qkv,
                kc + (size_t)l * BB * HH * MAXPOS * HDIM,
                vc + (size_t)l * BB * HH * MAXPOS * HDIM,
                attno, i + 1);
            launch_gemm(attno, h_saWo + (size_t)l * DD * DD,
                        sa_bo + (size_t)l * DD, tbuf, BB, DD, DD, 0, stream);
            cptd_resln<<<BB / 8, 256, 0, stream>>>(
                x, tbuf, ln1_g + (size_t)l * DD, ln1_b + (size_t)l * DD, BB);

            // cross-attention (precomputed) + LN2
            cptd_resln<<<BB / 8, 256, 0, stream>>>(
                x, ca_out + (size_t)l * BB * DD,
                ln2_g + (size_t)l * DD, ln2_b + (size_t)l * DD, BB);

            // FFN
            launch_gemm(x, h_lin1 + (size_t)l * 4 * DD * DD,
                        lin1_b + (size_t)l * 4 * DD, hbuf, BB, 4 * DD, DD, 1, stream);
            launch_gemm(hbuf, h_lin2 + (size_t)l * DD * 4 * DD,
                        lin2_b + (size_t)l * DD, tbuf, BB, DD, 4 * DD, 0, stream);
            cptd_resln<<<BB / 8, 256, 0, stream>>>(
                x, tbuf, ln3_g + (size_t)l * DD, ln3_b + (size_t)l * DD, BB);
        }

        hipMemcpyAsync(outs + (size_t)i * BB * DD, x, (size_t)BB * DD * sizeof(float),
                       hipMemcpyDeviceToDevice, stream);
    }

    // ---- output head
    cptd_out_proj<<<(BB * MAXPOS * 4 + 255) / 256, 256, 0, stream>>>(
        outs, out_W, out_b, plow, phigh, (float*)d_out);
}
